// BatchWiseTripletLoss_33663953666871
// MI455X (gfx1250) — matrix operand order repaired and checked by reading, not compiled, not run on hardware
//
#include <hip/hip_runtime.h>
#include <math.h>

// BatchWiseTripletLoss for MI455X (gfx1250, wave32).
//
// Pipeline (all on `stream`, deterministic, no fp atomics):
//   1) normalize_kernel : row L2-normalize emb (fp32) -> xh (f16) in ws
//   2) gemm_sim_kernel  : sim = xh @ xh^T via v_wmma_f32_16x16x32_f16.
//                         Each wave owns a 16x64 strip (4 tiles sharing the
//                         A fragment -> 4 WMMAs per 10 b128 loads). K-loop is
//                         2x unrolled with ping-pong register buffers so each
//                         buffer is loaded directly into its WMMA operand
//                         registers (no v_mov shuffling) and loads for the
//                         other buffer stay in flight behind the WMMAs.
//   3) row_loss_kernel  : per row, exact radix-select of the kept-th
//                         smallest negative (4x 8-bit histogram passes over
//                         the row cached in LDS), then masked pos/neg sums
//   4) final_reduce     : fixed-order tree sum of per-row losses, /n
//
// Workspace: xh 8MB | sim 64MB | per_row 16KB  (~72MB of d_ws)

typedef __attribute__((ext_vector_type(16))) _Float16 v16h;
typedef __attribute__((ext_vector_type(8)))  _Float16 v8h;
typedef __attribute__((ext_vector_type(8)))  float    v8f;

#define TL_MAXN 4096

// ---------------------------------------------------------------- utilities
__device__ __forceinline__ unsigned tl_fmap(float f) {
  // monotone float -> uint key (ascending float order == ascending uint order)
  unsigned u = __float_as_uint(f);
  return (u & 0x80000000u) ? ~u : (u | 0x80000000u);
}
__device__ __forceinline__ float tl_funmap(unsigned u) {
  unsigned v = (u & 0x80000000u) ? (u & 0x7FFFFFFFu) : ~u;
  return __uint_as_float(v);
}

// ------------------------------------------------------------ 1) normalize
__global__ __launch_bounds__(256) void tl_normalize_kernel(
    const float* __restrict__ emb, _Float16* __restrict__ xh, int d) {
  const int row = blockIdx.x;
  const int tid = threadIdx.x;
  __shared__ float red[256];
  const float* e = emb + (size_t)row * d;

  float ss = 0.0f;
  for (int j = tid; j < d; j += 256) { float v = e[j]; ss += v * v; }
  red[tid] = ss;
  __syncthreads();
  for (int st = 128; st > 0; st >>= 1) {
    if (tid < st) red[tid] += red[tid + st];
    __syncthreads();
  }
  const float inv = 1.0f / fmaxf(sqrtf(red[0]), 1e-12f);

  _Float16* o = xh + (size_t)row * d;
  for (int j = tid; j < d; j += 256) o[j] = (_Float16)(e[j] * inv);
}

// ---------------------------------------------- 2) sim = X X^T via WMMA f16
// Fragment layouts (16-bit, wave32):
//   A 16x32: lane L holds row M = L&15; halves 0-7 at K = 16*(L>>4),
//            halves 8-15 at K = 16*(L>>4) + 8.
//   B 32x16: lane L holds column N = L&15 (= row N of X for X*X^T),
//            16 contiguous K at base 16*(L>>4).
//   C:       VGPR j holds element (M = j + 8*(lane>>4), N = lane&15).
// Each wave computes 4 adjacent N-tiles (16x64 strip) reusing one A fragment.
// Requires d % 64 == 0 (d = 1024 here).

#define TL_LOAD_FRAGS(k, alo, ahi, b0, b1, b2, b3)    \
  do {                                                \
    alo = *(const v8h*)(arow + (k));                  \
    ahi = *(const v8h*)(arow + (k) + 8);              \
    b0  = *(const v16h*)(brow + (k));                 \
    b1  = *(const v16h*)(brow + (k) + bstride);       \
    b2  = *(const v16h*)(brow + (k) + 2 * bstride);   \
    b3  = *(const v16h*)(brow + (k) + 3 * bstride);   \
  } while (0)

#define TL_WMMA4(alo, ahi, b0, b1, b2, b3)                                  \
  do {                                                                      \
    v16h a = __builtin_shufflevector(alo, ahi, 0, 1, 2, 3, 4, 5, 6, 7, 8,   \
                                     9, 10, 11, 12, 13, 14, 15);            \
    c0 = __builtin_amdgcn_wmma_f32_16x16x32_f16(false, a, false, b0,        \
                                                (short)0, c0, false, false);\
    c1 = __builtin_amdgcn_wmma_f32_16x16x32_f16(false, a, false, b1,        \
                                                (short)0, c1, false, false);\
    c2 = __builtin_amdgcn_wmma_f32_16x16x32_f16(false, a, false, b2,        \
                                                (short)0, c2, false, false);\
    c3 = __builtin_amdgcn_wmma_f32_16x16x32_f16(false, a, false, b3,        \
                                                (short)0, c3, false, false);\
  } while (0)

__global__ __launch_bounds__(256) void tl_gemm_sim_kernel(
    const _Float16* __restrict__ xh, float* __restrict__ simM, int n, int d) {
  const int lane  = threadIdx.x & 31;
  const int wave  = threadIdx.x >> 5;
  const int rowBase  = blockIdx.y * 16;
  const int colTile0 = (blockIdx.x * 8 + wave) * 4;   // 4 consecutive N tiles
  const int lhalf = lane >> 4;   // 0 or 1
  const int l16   = lane & 15;

  const _Float16* arow = xh + (size_t)(rowBase + l16) * d + 16 * lhalf;
  const _Float16* brow = xh + (size_t)(colTile0 * 16 + l16) * d + 16 * lhalf;
  const size_t bstride = (size_t)16 * d;   // 16 rows of X per N-tile

  v8f c0 = {}, c1 = {}, c2 = {}, c3 = {};

  // ping-pong fragment buffers: loads write straight into WMMA operand regs
  v8h  alo0, ahi0, alo1, ahi1;
  v16h b00, b01, b02, b03, b10, b11, b12, b13;

  TL_LOAD_FRAGS(0, alo0, ahi0, b00, b01, b02, b03);

  for (int k0 = 0; k0 < d; k0 += 64) {
    // odd-step loads go out before even-step WMMAs (k0+32 < d since d%64==0)
    TL_LOAD_FRAGS(k0 + 32, alo1, ahi1, b10, b11, b12, b13);
    if (k0 + 64 < d)  // streaming hint -> global_prefetch_b8
      __builtin_prefetch((const void*)(brow + k0 + 64), 0, 3);

    TL_WMMA4(alo0, ahi0, b00, b01, b02, b03);

    if (k0 + 64 < d)
      TL_LOAD_FRAGS(k0 + 64, alo0, ahi0, b00, b01, b02, b03);

    TL_WMMA4(alo1, ahi1, b10, b11, b12, b13);
  }

  const int m0 = rowBase + lhalf * 8;
  float* outp = simM + (size_t)m0 * n + colTile0 * 16 + l16;
#pragma unroll
  for (int j = 0; j < 8; ++j) {
    float* o = outp + (size_t)j * n;
    o[0]  = c0[j];
    o[16] = c1[j];
    o[32] = c2[j];
    o[48] = c3[j];
  }
}

// ------------------------------------ 3) per-row radix-select + masked sums
__global__ __launch_bounds__(256) void tl_row_loss_kernel(
    const float* __restrict__ simM, const int* __restrict__ target,
    float* __restrict__ per_row, int n) {
  const int row = blockIdx.x;
  const int tid = threadIdx.x;

  __shared__ float         s[TL_MAXN];       // row of sim (16 KB)
  __shared__ unsigned char negb[TL_MAXN];    // negative mask (4 KB)
  __shared__ int           hist[256];
  __shared__ float         redf[256];
  __shared__ float         redf2[256];
  __shared__ int           redi[256];
  __shared__ unsigned      sh_prefix;
  __shared__ int           sh_rem;
  __shared__ int           sh_K, sh_haspos;
  __shared__ float         sh_posmax;

  const int    tgt  = target[row];
  const float* srow = simM + (size_t)row * n;

  // pass 0: stage row into LDS, count negatives, positive max
  int   myK = 0, myHas = 0;
  float myMax = -INFINITY;
  for (int j = tid; j < n; j += 256) {
    float v  = srow[j];
    int  neg = (target[j] != tgt);
    s[j]     = v;
    negb[j]  = (unsigned char)neg;
    myK += neg;
    if (!neg && j != row) { myHas = 1; myMax = fmaxf(myMax, v); }
  }
  redi[tid] = myK; redf[tid] = myMax;
  __syncthreads();
  for (int st = 128; st > 0; st >>= 1) {
    if (tid < st) {
      redi[tid] += redi[tid + st];
      redf[tid]  = fmaxf(redf[tid], redf[tid + st]);
    }
    __syncthreads();
  }
  if (tid == 0) { sh_K = redi[0]; sh_posmax = redf[0]; }
  __syncthreads();
  redi[tid] = myHas;
  __syncthreads();
  for (int st = 128; st > 0; st >>= 1) {
    if (tid < st) redi[tid] |= redi[tid + st];
    __syncthreads();
  }
  if (tid == 0) sh_haspos = redi[0];
  __syncthreads();

  const int   K      = sh_K;
  const float posMax = sh_posmax;
  const int   hasPos = sh_haspos;

  if (K == 0) {  // degenerate (all same class) -- uniform across block
    if (tid == 0) per_row[row] = 0.0f;
    return;
  }

  int drop = (int)floorf((float)K * 0.05f);
  if (drop < 1) drop = 1;
  int kept = K - drop;
  if (kept < 1) kept = 1;

  // radix-select the kept-th smallest negative (exact bit pattern)
  unsigned prefix = 0;
  int kk = kept;
  for (int pass = 0; pass < 4; ++pass) {
    const int shift = 24 - pass * 8;
    hist[tid] = 0;
    __syncthreads();
    for (int j = tid; j < n; j += 256) {
      if (negb[j]) {
        unsigned key = tl_fmap(s[j]);
        if (pass == 0 || (key >> (shift + 8)) == prefix)
          atomicAdd(&hist[(key >> shift) & 255], 1);
      }
    }
    __syncthreads();
    if (tid == 0) {
      int c = 0, b = 0;
      for (b = 0; b < 256; ++b) {
        int h = hist[b];
        if (c + h >= kk) break;
        c += h;
      }
      if (b == 256) b = 255;
      sh_prefix = (prefix << 8) | (unsigned)b;
      sh_rem    = kk - c;
    }
    __syncthreads();
    prefix = sh_prefix;
    kk     = sh_rem;
    __syncthreads();
  }

  const unsigned tk     = prefix;
  const float    thresh = tl_funmap(tk);
  const float    lower  = fmaxf(0.6f, posMax) - 0.1f;
  const float    posT   = thresh + 0.1f;

  // final masked sums
  float myPos = 0.0f, myNeg = 0.0f;
  int   myLt  = 0;
  for (int j = tid; j < n; j += 256) {
    float v = s[j];
    if (negb[j]) {
      unsigned key = tl_fmap(v);
      if (key < tk) {
        myLt++;
        if (v > lower) myNeg += v;
      }
    } else if (j != row) {
      if (v < posT) myPos += 1.0f - v;
    }
  }
  redf[tid]  = myPos;
  redf2[tid] = myNeg;
  redi[tid]  = myLt;
  __syncthreads();
  for (int st = 128; st > 0; st >>= 1) {
    if (tid < st) {
      redf[tid]  += redf[tid + st];
      redf2[tid] += redf2[tid + st];
      redi[tid]  += redi[tid + st];
    }
    __syncthreads();
  }
  if (tid == 0) {
    float negLoss = redf2[0];
    int   ties    = kept - redi[0];           // duplicates at thresh kept by rank
    if (ties < 0) ties = 0;
    if (thresh > lower) negLoss += (float)ties * thresh;
    per_row[row] = hasPos ? (redf[0] + negLoss) : 0.0f;
  }
}

// ------------------------------------------------------- 4) final reduction
__global__ __launch_bounds__(256) void tl_final_reduce_kernel(
    const float* __restrict__ per_row, float* __restrict__ out, int n) {
  const int tid = threadIdx.x;
  __shared__ float red[256];
  float sum = 0.0f;
  for (int j = tid; j < n; j += 256) sum += per_row[j];  // fixed order
  red[tid] = sum;
  __syncthreads();
  for (int st = 128; st > 0; st >>= 1) {
    if (tid < st) red[tid] += red[tid + st];
    __syncthreads();
  }
  if (tid == 0) out[0] = red[0] / (float)n;
}

// -------------------------------------------------------------------- launch
extern "C" void kernel_launch(void* const* d_in, const int* in_sizes, int n_in,
                              void* d_out, int out_size, void* d_ws, size_t ws_size,
                              hipStream_t stream) {
  const float* emb    = (const float*)d_in[0];
  const int*   target = (const int*)d_in[1];   // harness passes integer inputs as int32
  const int n = in_sizes[1];                   // 4096
  const int d = in_sizes[0] / n;               // 1024

  char* ws = (char*)d_ws;
  _Float16* xh = (_Float16*)ws;                               // n*d*2   = 8 MB
  size_t off = ((size_t)n * d * sizeof(_Float16) + 255) & ~(size_t)255;
  float* simM = (float*)(ws + off);                           // n*n*4   = 64 MB
  off += ((size_t)n * n * sizeof(float) + 255) & ~(size_t)255;
  float* per_row = (float*)(ws + off);                        // n*4     = 16 KB

  tl_normalize_kernel<<<n, 256, 0, stream>>>(emb, xh, d);

  // 8 waves/block, each wave owns a 16x64 strip (4 N-tiles)
  dim3 g((n / 16) / (8 * 4), n / 16);          // (8, 256) for n=4096
  tl_gemm_sim_kernel<<<g, 256, 0, stream>>>(xh, simM, n, d);

  tl_row_loss_kernel<<<n, 256, 0, stream>>>(simM, target, per_row, n);

  tl_final_reduce_kernel<<<1, 256, 0, stream>>>(per_row, (float*)d_out, n);
}
